// CustomConvnd_34600256537497
// MI455X (gfx1250) — matrix operand order repaired and checked by reading, not compiled, not run on hardware
//
#include <hip/hip_runtime.h>
#include <hip/hip_bf16.h>

// CDNA5 (gfx1250) implicit-GEMM 3x3 conv using v_wmma_f32_16x16x32_f16.
// GEMM view: M = COUT(256), N = B*H*W(131072), K = CIN*9(2304).
// Block tile: 128(co) x 256(spatial), 512 threads = 16 waves, wave tile 64x32.
// Per ci-block(32): stage all 9 weight taps + one x halo tile (6 rows x 66
// cols) into LDS in WMMA fragment order; the 9-tap loop is then pure
// ds_load_b128 (immediate offsets) + WMMA. Staging is division-free and
// rolled enough to avoid VGPR spills.

typedef __attribute__((ext_vector_type(16))) _Float16 v16h;
typedef __attribute__((ext_vector_type(8)))  float    v8f;

#define CONV_B    32
#define CONV_CIN  256
#define CONV_COUT 256
#define CONV_H    64
#define CONV_W    64
#define CONV_HW   (CONV_H * CONV_W)   // 4096

#define HALO_COLS (6 * 66)            // 396 halo columns (rh 0..5, wi 0..65)
#define SX_STRIDE 48                  // halfs per halo column (96B, 32B-aligned)

__global__ __launch_bounds__(512)
void conv3x3_wmma_f16(const float* __restrict__ x,
                      const float* __restrict__ wgt,
                      const float* __restrict__ bias,
                      float* __restrict__ out) {
    // A tiles for all 9 taps, WMMA A-fragment order:
    //   sA[pos*4096 + t*512 + lane*16 + halfIdx]
    __shared__ __align__(32) _Float16 sA[9 * 128 * 32];          // 72 KB
    // x halo tile, column-major: column ch = rh*66+wi holds 32 ci halfs.
    __shared__ __align__(32) _Float16 sX[HALO_COLS * SX_STRIDE]; // ~37 KB

    const int tid  = threadIdx.x;
    const int wv   = tid >> 5;
    const int lane = tid & 31;
    const int lane_lo = lane & 15;
    const int hi_half = (lane >= 16) ? 1 : 0;

    const int co_base = blockIdx.y * 128;
    const int s_base  = blockIdx.x * 256;          // 256 contiguous (n,h,w)
    const int n   = s_base >> 12;                  // / 4096
    const int hw0 = s_base & 4095;
    const int h0  = hw0 >> 6;                      // output rows h0..h0+3

    const int wv_co = wv & 1;                      // co sub-block of 64
    const int wv_sp = wv >> 1;                     // 0..7 -> spatial block of 32

    // Weight staging constants (division-free): 4 threads per co row.
    const int m_row  = tid >> 2;                   // 0..127 (co row in tile)
    const int lowq   = tid & 3;                    // k = lowq*8 + k8
    const int lA_st  = (m_row & 15) + ((lowq & 2) ? 16 : 0);
    const int hi8_st = (lowq & 1) ? 8 : 0;
    const int sA_base_st = (m_row >> 4) * 512 + lA_st * 16 + hi8_st;

    // x staging constants: rowid uniform per wave -> uniform h-bounds branch.
    const int rowid = tid >> 6;                    // 0..7
    const int wcol  = tid & 63;                    // 0..63 (valid w, coalesced)

    // Per-wave B-fragment column bases (tap (0,0)); tap adds kh*66+kw.
    int ch0[2];
#pragma unroll
    for (int j = 0; j < 2; ++j) {
        const int sl = wv_sp * 32 + j * 16 + lane_lo;  // 0..255
        ch0[j] = (sl >> 6) * 66 + (sl & 63);
    }

    // Edge halo columns (wi=0 -> w'=-1, wi=65 -> w'=64) are always zero;
    // write them once (first __syncthreads in the loop publishes them).
    if (tid < 384) {
        const int rh = tid >> 6;                   // 0..5
        const int r  = tid & 63;
        const int wi = (r >> 5) * 65;              // 0 or 65
        const int ci = r & 31;
        sX[(rh * 66 + wi) * SX_STRIDE + ci] = (_Float16)0.f;
    }

    v8f acc[4][2];
#pragma unroll
    for (int i = 0; i < 4; ++i)
#pragma unroll
        for (int j = 0; j < 2; ++j)
            acc[i][j] = (v8f){0.f, 0.f, 0.f, 0.f, 0.f, 0.f, 0.f, 0.f};

    for (int cb = 0; cb < 8; ++cb) {               // ci blocks of 32
        const int ci0 = cb * 32;

        // ---- Stage A: 9 taps, 128co x 32ci, fp32 -> f16, fragment order ----
        // Rolled over k8 (9 live loads max, no spill); taps unrolled so
        // pos*4096 stays an immediate ds offset.
        {
            const float* wrow = wgt
                + (size_t)(co_base + m_row) * (CONV_CIN * 9)
                + (size_t)(ci0 + lowq * 8) * 9;
#pragma unroll 1
            for (int k8 = 0; k8 < 8; ++k8) {
#pragma unroll
                for (int pos = 0; pos < 9; ++pos) {
                    const float wf = wrow[pos];
                    sA[pos * 4096 + (sA_base_st + k8)] = (_Float16)wf;
                }
                wrow += 9;
            }
        }

        // ---- Stage X halo interior: 32ci x 6rows x 64w, division-free ----
        // rid = rowid + 8e covers 0..191: rh = rid>>5, ci = rid&31.
        {
            const float* xb = x + ((size_t)n * CONV_CIN + ci0) * CONV_HW;
#pragma unroll 4
            for (int e = 0; e < 24; ++e) {
                const int rid = rowid + e * 8;
                const int rh  = rid >> 5;          // 0..5 (wave-uniform)
                const int ci  = rid & 31;          // wave-uniform
                const int hr  = h0 + rh - 1;
                float xv = 0.f;
                if (hr >= 0 && hr < CONV_H)        // uniform branch per wave
                    xv = xb[(size_t)ci * CONV_HW + hr * CONV_W + wcol];
                sX[(rh * 66 + wcol + 1) * SX_STRIDE + ci] = (_Float16)xv;
            }
        }

        __syncthreads();

        // ---- 9 taps: pure LDS fragment loads + WMMA ----
#pragma unroll
        for (int kh = 0; kh < 3; ++kh) {
#pragma unroll
            for (int kw = 0; kw < 3; ++kw) {
                const int pos = kh * 3 + kw;
                v16h aF[4], bF[2];
#pragma unroll
                for (int i = 0; i < 4; ++i) {
                    const int t = wv_co * 4 + i;
                    aF[i] = *(const v16h*)&sA[pos * 4096 + t * 512 + lane * 16];
                }
#pragma unroll
                for (int j = 0; j < 2; ++j) {
                    const int ch = ch0[j] + kh * 66 + kw;
                    bF[j] = *(const v16h*)&sX[ch * SX_STRIDE + hi_half * 16];
                }
#pragma unroll
                for (int i = 0; i < 4; ++i)
#pragma unroll
                    for (int j = 0; j < 2; ++j)
                        acc[i][j] = __builtin_amdgcn_wmma_f32_16x16x32_f16(
                            false, aF[i], false, bF[j],
                            (short)0, acc[i][j], false, false);
            }
        }

        __syncthreads();
    }

    // ---- Epilogue: bias + store (C/D layout: VGPR r -> M=r | M=8+r) ----
    const size_t out_n_base = (size_t)n * CONV_COUT * CONV_HW;
#pragma unroll
    for (int i = 0; i < 4; ++i) {
        const int co = co_base + (wv_co * 4 + i) * 16 + hi_half * 8;
#pragma unroll
        for (int j = 0; j < 2; ++j) {
            const int hw = hw0 + wv_sp * 32 + j * 16 + lane_lo;
#pragma unroll
            for (int r = 0; r < 8; ++r) {
                const int cor = co + r;
                out[out_n_base + (size_t)cor * CONV_HW + hw] =
                    acc[i][j][r] + bias[cor];
            }
        }
    }
}

extern "C" void kernel_launch(void* const* d_in, const int* in_sizes, int n_in,
                              void* d_out, int out_size, void* d_ws, size_t ws_size,
                              hipStream_t stream) {
    const float* x    = (const float*)d_in[0];   // (32, 256, 64, 64)
    const float* wgt  = (const float*)d_in[1];   // (256, 256, 3, 3)
    const float* bias = (const float*)d_in[2];   // (256,)
    float* out = (float*)d_out;                  // (32, 256, 64, 64)

    dim3 grid((CONV_B * CONV_HW) / 256, CONV_COUT / 128);  // (512, 2)
    dim3 block(512);
    conv3x3_wmma_f16<<<grid, block, 0, stream>>>(x, wgt, bias, out);
}
